// MPSTemporalBlock_82497731821673
// MI455X (gfx1250) — compile-verified, matrix operand order checked by reading, not run on hardware
//
#include <hip/hip_runtime.h>
#include <hip/hip_bf16.h>
#include <math.h>

// ---------------- CDNA5 WMMA types ----------------
typedef __attribute__((ext_vector_type(16))) __bf16 v16bf;
typedef __attribute__((ext_vector_type(8)))  float  v8f;

union Frag {
    v16bf v;
    uint4 q[2];
};

// Model dims
#define Bq   4
#define Lq   1024
#define Dq   512
#define CHI  32
#define Sq   4
#define ROWS (Bq * Lq)          // 4096
#define NSITE (CHI * Sq * CHI)  // 4096
#define SCOL (Sq * CHI)         // 128

__device__ __forceinline__ unsigned short f2bf(float f) {
    unsigned int u = __float_as_uint(f);
    unsigned int r = u + 0x7FFFu + ((u >> 16) & 1u);  // RNE to bf16
    return (unsigned short)(r >> 16);
}

// -------- gfx1250 async global->LDS copy (ASYNCcnt path) --------
// Copies 16 KB: 32 chunks x (32 lanes x 16B). VDST = LDS byte address VGPR,
// VADDR = 64-bit global address (GV mode, saddr = off).
__device__ __forceinline__ void async_copy_tile16k(const float* __restrict__ g,
                                                   float* lbuf, int lane) {
    unsigned base_l = (unsigned)(uintptr_t)lbuf;
    const char* gp = (const char*)g;
    #pragma unroll
    for (int c = 0; c < 32; ++c) {
        unsigned    loff = base_l + (unsigned)(c * 512 + lane * 16);
        const void* ga   = gp + c * 512 + lane * 16;
        asm volatile("global_load_async_to_lds_b128 %0, %1, off"
                     :: "v"(loff), "v"(ga) : "memory");
    }
}

__device__ __forceinline__ void wait_async0() {
    asm volatile("s_wait_asynccnt 0" ::: "memory");
}

__device__ __forceinline__ float wave_allreduce_add(float v) {
    #pragma unroll
    for (int m = 1; m < 32; m <<= 1) v += __shfl_xor(v, m, 32);
    return v;
}

// =====================================================================
// GEMM: C[M,N] = A[M,K] @ Bw[K,N] + bias[N]   (bf16 WMMA, f32 accum)
// EPI==0: bias only; EPI==1: sigmoid(.+bias)
// Block: 256 threads (8 waves). Tile: 64x64, K-step 32.
// =====================================================================
template <int EPI>
__global__ void gemm_bias_kernel(const float* __restrict__ A,
                                 const float* __restrict__ Bw,
                                 const float* __restrict__ bias,
                                 float* __restrict__ C,
                                 int M, int N, int K) {
    const int tid  = threadIdx.x;
    const int lane = tid & 31;
    const int w    = tid >> 5;
    const int m0 = blockIdx.y * 64;
    const int n0 = blockIdx.x * 64;

    __shared__ __align__(16) unsigned short smA[64 * 40];
    __shared__ __align__(16) unsigned short smB[64 * 40];

    const int mi    = w & 3;
    const int nbase = (w >> 2) * 2;

    v8f acc0 = {0.f,0.f,0.f,0.f,0.f,0.f,0.f,0.f};
    v8f acc1 = {0.f,0.f,0.f,0.f,0.f,0.f,0.f,0.f};

    for (int k0 = 0; k0 < K; k0 += 32) {
        for (int i = tid; i < 64 * 32; i += 256) {
            int r = i >> 5, k = i & 31;
            smA[r * 40 + k] = f2bf(A[(size_t)(m0 + r) * K + k0 + k]);
        }
        for (int i = tid; i < 64 * 32; i += 256) {
            int n = i & 63, k = i >> 6;
            smB[n * 40 + k] = f2bf(Bw[(size_t)(k0 + k) * N + n0 + n]);
        }
        __syncthreads();

        const int klo  = (lane >= 16) ? 8 : 0;
        const int arow = mi * 16 + (lane & 15);
        Frag fa, fb0, fb1;
        fa.q[0] = *(const uint4*)&smA[arow * 40 + klo];
        fa.q[1] = *(const uint4*)&smA[arow * 40 + 16 + klo];
        const int c0 = (nbase + 0) * 16 + (lane & 15);
        fb0.q[0] = *(const uint4*)&smB[c0 * 40 + klo];
        fb0.q[1] = *(const uint4*)&smB[c0 * 40 + 16 + klo];
        const int c1 = (nbase + 1) * 16 + (lane & 15);
        fb1.q[0] = *(const uint4*)&smB[c1 * 40 + klo];
        fb1.q[1] = *(const uint4*)&smB[c1 * 40 + 16 + klo];

        acc0 = __builtin_amdgcn_wmma_f32_16x16x32_bf16(
            false, fa.v, false, fb0.v, (short)0, acc0, false, false);
        acc1 = __builtin_amdgcn_wmma_f32_16x16x32_bf16(
            false, fa.v, false, fb1.v, (short)0, acc1, false, false);
        __syncthreads();
    }

    const int mrow  = m0 + mi * 16 + ((lane >> 4) << 3);
    const int ncol0 = n0 + nbase * 16 + (lane & 15);
    #pragma unroll
    for (int e = 0; e < 8; ++e) {
        float v0 = acc0[e] + bias[ncol0];
        float v1 = acc1[e] + bias[ncol0 + 16];
        if (EPI == 1) {
            v0 = 1.f / (1.f + expf(-v0));
            v1 = 1.f / (1.f + expf(-v1));
        }
        C[(size_t)(mrow + e) * N + ncol0]      = v0;
        C[(size_t)(mrow + e) * N + ncol0 + 16] = v1;
    }
}

// =====================================================================
// W_comb[128,512] = W_bridge @ W_out ; b_comb = b_bridge @ W_out + b_out
// =====================================================================
__global__ void comb_kernel(const float* __restrict__ W_bridge,
                            const float* __restrict__ b_bridge,
                            const float* __restrict__ W_out,
                            const float* __restrict__ b_out,
                            float* __restrict__ W_comb,
                            float* __restrict__ b_comb) {
    int i = blockIdx.x * 256 + threadIdx.x;
    if (i < SCOL * Dq) {
        int p = i >> 9, n = i & 511;
        float s = 0.f;
        #pragma unroll
        for (int k = 0; k < CHI; ++k)
            s += W_bridge[p * CHI + k] * W_out[k * Dq + n];
        W_comb[i] = s;
    } else if (i < SCOL * Dq + Dq) {
        int n = i - SCOL * Dq;
        float s = b_out[n];
        #pragma unroll
        for (int k = 0; k < CHI; ++k)
            s += b_bridge[k] * W_out[k * Dq + n];
        b_comb[n] = s;
    }
}

// =====================================================================
// Barrier-free MPS scan: one wave32 per batch chain.
// Lane r owns column r of M[4][32]. Async double-buffered LDS staging
// of the 16 KB site tensor per step; Gram reduced via shfl_xor so every
// lane holds G; Gershgorin fast-path skips the Jacobi SVD when no
// singular value can fall below 1e-4 * sigma_max.
// =====================================================================
__global__ __launch_bounds__(32)
void scan_wave_kernel(const float* __restrict__ x,
                      const float* __restrict__ site,   // [4096][4096]
                      float* __restrict__ outs) {       // [4096][128]
    const int lane = threadIdx.x;    // 0..31
    const int b    = blockIdx.x;

    __shared__ __align__(16) float bufA[2][NSITE];      // 2 x 16 KB

    // preload t = 0
    async_copy_tile16k(site + (size_t)b * Lq * NSITE, &bufA[0][0], lane);

    float hv = 0.f;

    for (int t = 0; t < Lq; ++t) {
        const size_t row = (size_t)b * Lq + t;
        const int cur = t & 1, nxt = cur ^ 1;

        wait_async0();                       // buffer `cur` is complete
        if (t + 1 < Lq) {                    // stream t+1 during compute
            async_copy_tile16k(site + (row + 1) * NSITE, &bufA[nxt][0], lane);
            __builtin_prefetch(x + (row + 1) * Dq + lane, 0, 3);
        }

        // left = h + x[b,t,0:32]  (lane l holds left[l])
        float leftv = hv + x[row * Dq + lane];

        // M[p][r] = sum_l left[l] * A[l][p][r]
        const float* Ac = &bufA[cur][0];
        float m0 = 0.f, m1 = 0.f, m2 = 0.f, m3 = 0.f;
        #pragma unroll 8
        for (int l = 0; l < CHI; ++l) {
            float ls = __shfl(leftv, l, 32);
            m0 = fmaf(ls, Ac[l * SCOL +       lane], m0);
            m1 = fmaf(ls, Ac[l * SCOL +  32 + lane], m1);
            m2 = fmaf(ls, Ac[l * SCOL +  64 + lane], m2);
            m3 = fmaf(ls, Ac[l * SCOL +  96 + lane], m3);
        }

        // Gram G = M M^T (all lanes end up with the full 4x4)
        float g00 = wave_allreduce_add(m0 * m0);
        float g01 = wave_allreduce_add(m0 * m1);
        float g02 = wave_allreduce_add(m0 * m2);
        float g03 = wave_allreduce_add(m0 * m3);
        float g11 = wave_allreduce_add(m1 * m1);
        float g12 = wave_allreduce_add(m1 * m2);
        float g13 = wave_allreduce_add(m1 * m3);
        float g22 = wave_allreduce_add(m2 * m2);
        float g23 = wave_allreduce_add(m2 * m3);
        float g33 = wave_allreduce_add(m3 * m3);

        // Gershgorin sufficient condition: lambda_min >= 1e-8 * lambda_max
        float tr = g00 + g11 + g22 + g33;    // >= lambda_max
        float r0 = g00 - (fabsf(g01) + fabsf(g02) + fabsf(g03));
        float r1 = g11 - (fabsf(g01) + fabsf(g12) + fabsf(g13));
        float r2 = g22 - (fabsf(g02) + fabsf(g12) + fabsf(g23));
        float r3 = g33 - (fabsf(g03) + fabsf(g13) + fabsf(g23));
        float lo = fminf(fminf(r0, r1), fminf(r2, r3));

        float t0, t1, t2, t3;                 // Mt columns (this lane)
        if (lo >= 1e-8f * tr) {
            // no truncation possible: Mt = M
            t0 = m0; t1 = m1; t2 = m2; t3 = m3;
        } else {
            // per-lane redundant 4x4 Jacobi (uniform data / uniform branch)
            float Am[4][4] = {{g00,g01,g02,g03},{g01,g11,g12,g13},
                              {g02,g12,g22,g23},{g03,g13,g23,g33}};
            float Vm[4][4] = {{1,0,0,0},{0,1,0,0},{0,0,1,0},{0,0,0,1}};
            for (int sw = 0; sw < 6; ++sw)
                for (int pp = 0; pp < 3; ++pp)
                    for (int q = pp + 1; q < 4; ++q) {
                        float apq = Am[pp][q];
                        if (fabsf(apq) < 1e-30f) continue;
                        float tau = (Am[q][q] - Am[pp][pp]) / (2.f * apq);
                        float tt  = (tau >= 0.f ? 1.f : -1.f) /
                                    (fabsf(tau) + sqrtf(1.f + tau * tau));
                        float c = rsqrtf(1.f + tt * tt);
                        float s = tt * c;
                        for (int k = 0; k < 4; ++k) {
                            float akp = Am[k][pp], akq = Am[k][q];
                            Am[k][pp] = c * akp - s * akq;
                            Am[k][q]  = s * akp + c * akq;
                        }
                        for (int k = 0; k < 4; ++k) {
                            float apk = Am[pp][k], aqk = Am[q][k];
                            Am[pp][k] = c * apk - s * aqk;
                            Am[q][k]  = s * apk + c * aqk;
                        }
                        for (int k = 0; k < 4; ++k) {
                            float vkp = Vm[k][pp], vkq = Vm[k][q];
                            Vm[k][pp] = c * vkp - s * vkq;
                            Vm[k][q]  = s * vkp + c * vkq;
                        }
                    }
            float wv[4], wmax = 0.f;
            #pragma unroll
            for (int k = 0; k < 4; ++k) {
                wv[k] = fmaxf(Am[k][k], 0.f);
                wmax  = fmaxf(wmax, wv[k]);
            }
            const float thr = 1e-8f * wmax;
            float msk[4];
            #pragma unroll
            for (int k = 0; k < 4; ++k) msk[k] = (wv[k] >= thr) ? 1.f : 0.f;
            float P[4][4];
            #pragma unroll
            for (int i = 0; i < 4; ++i)
                #pragma unroll
                for (int j = 0; j < 4; ++j) {
                    float s = 0.f;
                    #pragma unroll
                    for (int k = 0; k < 4; ++k)
                        s += Vm[i][k] * Vm[j][k] * msk[k];
                    P[i][j] = s;
                }
            t0 = P[0][0]*m0 + P[0][1]*m1 + P[0][2]*m2 + P[0][3]*m3;
            t1 = P[1][0]*m0 + P[1][1]*m1 + P[1][2]*m2 + P[1][3]*m3;
            t2 = P[2][0]*m0 + P[2][1]*m1 + P[2][2]*m2 + P[2][3]*m3;
            t3 = P[3][0]*m0 + P[3][1]*m1 + P[3][2]*m2 + P[3][3]*m3;
        }

        // outputs
        float* ob = outs + row * SCOL + lane;
        ob[0]  = t0;
        ob[32] = t1;
        ob[64] = t2;
        ob[96] = t3;

        // h = v / (||v|| + eps),  v = sum_p Mt[p][:]
        float v  = t0 + t1 + t2 + t3;
        float ss = wave_allreduce_add(v * v);
        hv = v / (sqrtf(ss) + 1e-12f);
    }
}

// =====================================================================
// Fused: y = outs @ W_comb + b_comb ; LayerNorm ; gated residual.
// Block: 16 rows x 512 cols; ysh aliases smB staging (dead after K loop).
// =====================================================================
__global__ void out_fused_kernel(const float* __restrict__ outs,
                                 const float* __restrict__ W_comb,
                                 const float* __restrict__ b_comb,
                                 const float* __restrict__ gate,
                                 const float* __restrict__ x,
                                 const float* __restrict__ gamma,
                                 const float* __restrict__ beta,
                                 float* __restrict__ out) {
    const int tid  = threadIdx.x;
    const int lane = tid & 31;
    const int w    = tid >> 5;
    const int m0   = blockIdx.x * 16;

    __shared__ __align__(16) unsigned short smA[16 * 40];
    __shared__ __align__(16) unsigned char  smU[512 * 40 * 2]; // 40 KB
    unsigned short* smB = (unsigned short*)smU;
    float*          ysh = (float*)smU;                          // [16][512]
    __shared__ float part1[16][16], part2[16][16];
    __shared__ float muL[16], invL[16];

    v8f acc[4];
    #pragma unroll
    for (int j = 0; j < 4; ++j)
        acc[j] = (v8f){0.f,0.f,0.f,0.f,0.f,0.f,0.f,0.f};

    for (int k0 = 0; k0 < SCOL; k0 += 32) {
        for (int i = tid; i < 16 * 32; i += 256) {
            int r = i >> 5, k = i & 31;
            smA[r * 40 + k] = f2bf(outs[(size_t)(m0 + r) * SCOL + k0 + k]);
        }
        for (int i = tid; i < 512 * 32; i += 256) {
            int n = i & 511, k = i >> 9;
            smB[n * 40 + k] = f2bf(W_comb[(size_t)(k0 + k) * Dq + n]);
        }
        __syncthreads();

        const int klo  = (lane >= 16) ? 8 : 0;
        const int arow = lane & 15;
        Frag fa;
        fa.q[0] = *(const uint4*)&smA[arow * 40 + klo];
        fa.q[1] = *(const uint4*)&smA[arow * 40 + 16 + klo];
        #pragma unroll
        for (int j = 0; j < 4; ++j) {
            const int col = (w * 4 + j) * 16 + (lane & 15);
            Frag fb;
            fb.q[0] = *(const uint4*)&smB[col * 40 + klo];
            fb.q[1] = *(const uint4*)&smB[col * 40 + 16 + klo];
            acc[j] = __builtin_amdgcn_wmma_f32_16x16x32_bf16(
                false, fa.v, false, fb.v, (short)0, acc[j], false, false);
        }
        __syncthreads();
    }

    {
        const int mbase = (lane >> 4) << 3;
        #pragma unroll
        for (int j = 0; j < 4; ++j) {
            const int col = (w * 4 + j) * 16 + (lane & 15);
            const float bc = b_comb[col];
            #pragma unroll
            for (int e = 0; e < 8; ++e)
                ysh[(mbase + e) * Dq + col] = acc[j][e] + bc;
        }
    }
    __syncthreads();

    {
        int m = tid >> 4, c0 = tid & 15;
        float s1 = 0.f, s2 = 0.f;
        for (int c = c0; c < Dq; c += 16) {
            float v = ysh[m * Dq + c];
            s1 += v;
            s2 += v * v;
        }
        part1[m][c0] = s1;
        part2[m][c0] = s2;
    }
    __syncthreads();
    if (tid < 16) {
        float s1 = 0.f, s2 = 0.f;
        #pragma unroll
        for (int k = 0; k < 16; ++k) { s1 += part1[tid][k]; s2 += part2[tid][k]; }
        float mean = s1 * (1.f / Dq);
        float var  = s2 * (1.f / Dq) - mean * mean;
        muL[tid]  = mean;
        invL[tid] = rsqrtf(var + 1e-6f);
    }
    __syncthreads();

    for (int i = tid; i < 16 * Dq; i += 256) {
        int m = i >> 9, col = i & 511;
        size_t row = (size_t)(m0 + m);
        float yn = (ysh[i] - muL[m]) * invL[m] * gamma[col] + beta[col];
        float g  = gate[row * Dq + col];
        float xv = x[row * Dq + col];
        out[row * Dq + col] = g * yn + (1.f - g) * xv;
    }
}

// =====================================================================
extern "C" void kernel_launch(void* const* d_in, const int* in_sizes, int n_in,
                              void* d_out, int out_size, void* d_ws, size_t ws_size,
                              hipStream_t stream) {
    const float* x        = (const float*)d_in[0];
    const float* W_site   = (const float*)d_in[1];
    const float* b_site   = (const float*)d_in[2];
    const float* W_bridge = (const float*)d_in[3];
    const float* b_bridge = (const float*)d_in[4];
    const float* W_out    = (const float*)d_in[5];
    const float* b_out    = (const float*)d_in[6];
    const float* gamma    = (const float*)d_in[7];
    const float* beta     = (const float*)d_in[8];
    const float* W_gate   = (const float*)d_in[9];
    const float* b_gate   = (const float*)d_in[10];
    float* out = (float*)d_out;

    char* ws = (char*)d_ws;
    float* ws_site = (float*)(ws);                                  // 64 MB
    float* ws_gate = (float*)(ws + (size_t)ROWS * NSITE * 4);       //  8 MB
    float* ws_outs = (float*)(ws + (size_t)ROWS * NSITE * 4
                                 + (size_t)ROWS * Dq * 4);          //  2 MB
    float* ws_Wc   = (float*)(ws + (size_t)ROWS * NSITE * 4
                                 + (size_t)ROWS * Dq * 4
                                 + (size_t)ROWS * SCOL * 4);        // 256 KB
    float* ws_bc   = ws_Wc + (size_t)SCOL * Dq;                     //   2 KB

    gemm_bias_kernel<0><<<dim3(NSITE / 64, ROWS / 64), 256, 0, stream>>>(
        x, W_site, b_site, ws_site, ROWS, NSITE, Dq);

    gemm_bias_kernel<1><<<dim3(Dq / 64, ROWS / 64), 256, 0, stream>>>(
        x, W_gate, b_gate, ws_gate, ROWS, Dq, Dq);

    comb_kernel<<<(SCOL * Dq + Dq + 255) / 256, 256, 0, stream>>>(
        W_bridge, b_bridge, W_out, b_out, ws_Wc, ws_bc);

    scan_wave_kernel<<<Bq, 32, 0, stream>>>(x, ws_site, ws_outs);

    out_fused_kernel<<<ROWS / 16, 256, 0, stream>>>(
        ws_outs, ws_Wc, ws_bc, ws_gate, x, gamma, beta, out);
}
